// LearnablePrompts_36069135352245
// MI455X (gfx1250) — matrix operand order repaired and checked by reading, not compiled, not run on hardware
//
#include <hip/hip_runtime.h>
#include <hip/hip_bf16.h>

// ---------------------------------------------------------------------------
// Cosine-similarity argmax (embeddings [P,D] x table [V,D]) for gfx1250.
// Split-bf16 (hi+lo) WMMA GEMM: dot ~= hi*hi + hi*lo + lo*hi  (fp32-class
// accuracy at v_wmma_f32_16x16x32_bf16 speed).  e-norm drops out of argmax;
// t-norm applied in the epilogue before a u64 atomic-max argmax.
// Tiles are streamed L2->LDS with GLOBAL_LOAD_ASYNC_TO_LDS_B128 through a
// 3-deep ring buffer (one barrier per K-step, ASYNCcnt-pipelined).
// ---------------------------------------------------------------------------

typedef __bf16 bf16_t;
typedef bf16_t v16bf __attribute__((ext_vector_type(16)));
typedef float  v8f   __attribute__((ext_vector_type(8)));

union FragBF { uint4 u[2]; v16bf v; };

#define TILE_M 128
#define TILE_N 128
#define TILE_K 32
#define LSTR   40            // LDS row stride in bf16 elems (80B: 16B-aligned, conflict-free)
#define NBUF   3             // async ring depth (120KB LDS/block -> 2 blocks/WGP)
#define GEMM_THREADS 512     // 16 waves: 8 (M) x 2 (N); each wave = 16x64 output

// Async L2->LDS copy of one 16B chunk per lane (VGLOBAL GVS addressing).
__device__ __forceinline__ void async_load16(unsigned ldsAddr,
                                             unsigned long long gbase,
                                             unsigned goff) {
  asm volatile("global_load_async_to_lds_b128 %0, %1, %2"
               :: "v"(ldsAddr), "v"(goff), "s"(gbase) : "memory");
}
#define WAIT_ASYNC_4() asm volatile("s_wait_asynccnt 0x4" ::: "memory")
#define WAIT_ASYNC_0() asm volatile("s_wait_asynccnt 0x0" ::: "memory")

// ---------------------------------------------------------------------------
// Pass 1: fp32 row -> (hi, lo) bf16 split; optional inv-norm output.
// Rows >= nrowsReal are zero-filled padding (inv = 0).
// ---------------------------------------------------------------------------
__global__ __launch_bounds__(256)
void convert_rows(const float* __restrict__ src,
                  bf16_t* __restrict__ hi, bf16_t* __restrict__ lo,
                  float* __restrict__ inv, int nrowsReal, int D)
{
  const int  row  = blockIdx.x;
  const int  t    = threadIdx.x;
  const long base = (long)row * D;

  if (row >= nrowsReal) {
    for (int i = t; i < D; i += 256) {
      hi[base + i] = (bf16_t)0.0f;
      lo[base + i] = (bf16_t)0.0f;
    }
    if (inv && t == 0) inv[row] = 0.0f;
    return;
  }

  float ss = 0.0f;
  for (int i = t; i < D; i += 256) {
    float  x = src[base + i];
    bf16_t h = (bf16_t)x;              // RNE f32->bf16
    float  r = x - (float)h;           // exact residual
    bf16_t l = (bf16_t)r;
    hi[base + i] = h;
    lo[base + i] = l;
    ss += x * x;
  }

  if (inv) {
    __shared__ float red[256];
    red[t] = ss;
    __syncthreads();
    for (int s = 128; s > 0; s >>= 1) {
      if (t < s) red[t] += red[t + s];
      __syncthreads();
    }
    if (t == 0) inv[row] = 1.0f / fmaxf(sqrtf(red[0]), 1e-8f);
  }
}

// ---------------------------------------------------------------------------
// Zero the per-row argmax keys (ws is poisoned; must re-init every launch).
// ---------------------------------------------------------------------------
__global__ void init_keys(unsigned long long* __restrict__ keys, int P)
{
  int i = blockIdx.x * blockDim.x + threadIdx.x;
  if (i < P) keys[i] = 0ULL;
}

// ---------------------------------------------------------------------------
// Pass 2: 128x128x4096 split-bf16 WMMA GEMM with fused scale + argmax.
// ---------------------------------------------------------------------------
__global__ __launch_bounds__(GEMM_THREADS)
void wmma_gemm_argmax(const bf16_t* __restrict__ eHi, const bf16_t* __restrict__ eLo,
                      const bf16_t* __restrict__ tHi, const bf16_t* __restrict__ tLo,
                      const float* __restrict__ invT,
                      unsigned long long* __restrict__ keys,
                      int D, int V)
{
  __shared__ bf16_t sAhi[NBUF][TILE_M * LSTR];
  __shared__ bf16_t sAlo[NBUF][TILE_M * LSTR];
  __shared__ bf16_t sBhi[NBUF][TILE_N * LSTR];
  __shared__ bf16_t sBlo[NBUF][TILE_N * LSTR];

  const int  tid   = threadIdx.x;
  const long pbase = (long)blockIdx.x * TILE_M;   // x fastest => 16 P-tiles reuse one V-tile in L2
  const long vbase = (long)blockIdx.y * TILE_N;

  // ---- tile loader: 512 threads x one 16B chunk per array per K-step ----
  const int lrow = tid >> 2;                  // 0..127
  const int lseg = (tid & 3) << 3;            // elem offset 0,8,16,24
  const unsigned gOffA = (unsigned)(((pbase + lrow) * (long)D + lseg) * 2);  // bytes
  const unsigned gOffB = (unsigned)(((vbase + lrow) * (long)D + lseg) * 2);  // bytes (< 412MB)
  const unsigned long long bAh = (unsigned long long)(uintptr_t)eHi;
  const unsigned long long bAl = (unsigned long long)(uintptr_t)eLo;
  const unsigned long long bBh = (unsigned long long)(uintptr_t)tHi;
  const unsigned long long bBl = (unsigned long long)(uintptr_t)tLo;
  const int lodst = lrow * LSTR + lseg;

  // per-buffer LDS byte addresses for this thread's chunk
  unsigned dAh[NBUF], dAl[NBUF], dBh[NBUF], dBl[NBUF];
  #pragma unroll
  for (int b = 0; b < NBUF; ++b) {
    dAh[b] = (unsigned)(uintptr_t)&sAhi[b][lodst];
    dAl[b] = (unsigned)(uintptr_t)&sAlo[b][lodst];
    dBh[b] = (unsigned)(uintptr_t)&sBhi[b][lodst];
    dBl[b] = (unsigned)(uintptr_t)&sBlo[b][lodst];
  }

  // ---- compute mapping (wave32) ----
  const int lane = tid & 31;
  const int wv   = tid >> 5;                 // 0..15
  const int wm   = wv >> 1;                  // 0..7  (M)
  const int wn   = wv & 1;                   // 0..1  (N)
  const int fr   = lane & 15;
  const int h    = lane >> 4;                // half-wave select
  // A frag (16x32 bf16, ISA 7.12.2): lane m=fr, VGPR0-3 = K[h*8 .. +7], VGPR4-7 = K[16+h*8 .. +7]
  const int aB = (wm * 16 + fr) * LSTR;
  const int a0 = aB + h * 8;
  const int a1 = aB + 16 + h * 8;
  // B frag (32x16 bf16): lane n=fr, VGPR0-3 = K[h*8 .. +7], VGPR4-7 = K[16+h*8 .. +7]
  int b0[4], b1[4];
  #pragma unroll
  for (int s = 0; s < 4; ++s) {
    int bB = (wn * 64 + s * 16 + fr) * LSTR;
    b0[s] = bB + h * 8;
    b1[s] = bB + 16 + h * 8;
  }

  v8f acc[4] = {};

  const int nk = D / TILE_K;

  // prologue: fill two ring slots
  {
    async_load16(dAh[0], bAh, gOffA);
    async_load16(dAl[0], bAl, gOffA);
    async_load16(dBh[0], bBh, gOffB);
    async_load16(dBl[0], bBl, gOffB);
    async_load16(dAh[1], bAh, gOffA + TILE_K * 2);
    async_load16(dAl[1], bAl, gOffA + TILE_K * 2);
    async_load16(dBh[1], bBh, gOffB + TILE_K * 2);
    async_load16(dBl[1], bBl, gOffB + TILE_K * 2);
  }

  int cur = 0;                                // ks % NBUF
  for (int ks = 0; ks < nk; ++ks) {
    // buffer `cur` loaded when only the newer slot's 4 ops remain (in-order).
    if (ks + 1 < nk) { WAIT_ASYNC_4(); } else { WAIT_ASYNC_0(); }
    __syncthreads();                          // all waves' data landed; compute(ks-1) done everywhere

    if (ks + 2 < nk) {                        // refill slot (ks+2)%NBUF (= (ks-1)%NBUF, now free)
      int nb = cur + 2; if (nb >= NBUF) nb -= NBUF;
      const unsigned kb = (unsigned)((ks + 2) * TILE_K * 2);
      async_load16(dAh[nb], bAh, gOffA + kb);
      async_load16(dAl[nb], bAl, gOffA + kb);
      async_load16(dBh[nb], bBh, gOffB + kb);
      async_load16(dBl[nb], bBl, gOffB + kb);
    }

    FragBF ah, al;
    ah.u[0] = *(const uint4*)&sAhi[cur][a0];
    ah.u[1] = *(const uint4*)&sAhi[cur][a1];
    al.u[0] = *(const uint4*)&sAlo[cur][a0];
    al.u[1] = *(const uint4*)&sAlo[cur][a1];
    #pragma unroll
    for (int s = 0; s < 4; ++s) {
      FragBF bh, bl;
      bh.u[0] = *(const uint4*)&sBhi[cur][b0[s]];
      bh.u[1] = *(const uint4*)&sBhi[cur][b1[s]];
      bl.u[0] = *(const uint4*)&sBlo[cur][b0[s]];
      bl.u[1] = *(const uint4*)&sBlo[cur][b1[s]];
      acc[s] = __builtin_amdgcn_wmma_f32_16x16x32_bf16(false, ah.v, false, bh.v,
                                                       (short)0, acc[s], false, false);
      acc[s] = __builtin_amdgcn_wmma_f32_16x16x32_bf16(false, ah.v, false, bl.v,
                                                       (short)0, acc[s], false, false);
      acc[s] = __builtin_amdgcn_wmma_f32_16x16x32_bf16(false, al.v, false, bh.v,
                                                       (short)0, acc[s], false, false);
    }

    if (++cur == NBUF) cur = 0;
  }

  // ---- epilogue: scale by inv_t[v], argmax (ties -> smaller v, like jnp.argmax) ----
  // C/D layout: lane column n = fr; VGPR g holds row M = g + 8*h.
  float bestv[8];
  int   besti[8];
  #pragma unroll
  for (int g = 0; g < 8; ++g) { bestv[g] = -3.0e38f; besti[g] = 0; }

  #pragma unroll
  for (int s = 0; s < 4; ++s) {
    const long vcol  = vbase + wn * 64 + s * 16 + fr;
    const float it   = invT[vcol];            // padded rows read 0, masked below
    const bool valid = vcol < (long)V;
    #pragma unroll
    for (int g = 0; g < 8; ++g) {
      float val = valid ? acc[s][g] * it : -3.0e38f;
      if (val > bestv[g] || (val == bestv[g] && (int)vcol < besti[g])) {
        bestv[g] = val; besti[g] = (int)vcol;
      }
    }
  }
  // reduce across the 16 lanes of each half-wave (they share the same 8 p-rows)
  #pragma unroll
  for (int m = 1; m < 16; m <<= 1) {
    #pragma unroll
    for (int g = 0; g < 8; ++g) {
      float ov = __shfl_xor(bestv[g], m, 32);
      int   oi = __shfl_xor(besti[g], m, 32);
      if (ov > bestv[g] || (ov == bestv[g] && oi < besti[g])) {
        bestv[g] = ov; besti[g] = oi;
      }
    }
  }
  if (fr == 0) {
    #pragma unroll
    for (int g = 0; g < 8; ++g) {
      const long p = pbase + wm * 16 + h * 8 + g;
      unsigned b   = __float_as_uint(bestv[g]);
      unsigned k32 = b ^ ((unsigned)((int)b >> 31) | 0x80000000u);  // monotonic f32 map
      unsigned long long key =
          ((unsigned long long)k32 << 32) | (unsigned)(~(unsigned)besti[g]); // ~v: ties->smaller v
      atomicMax(&keys[p], key);
    }
  }
}

// ---------------------------------------------------------------------------
// Pass 3: decode keys -> float ids.
// ---------------------------------------------------------------------------
__global__ void finalize_ids(const unsigned long long* __restrict__ keys,
                             float* __restrict__ out, int P)
{
  int p = blockIdx.x * blockDim.x + threadIdx.x;
  if (p < P) {
    unsigned idx = ~(unsigned)(keys[p] & 0xFFFFFFFFu);
    out[p] = (float)idx;
  }
}

// ---------------------------------------------------------------------------
// Fallback (ws_size too small, or P/D not tile multiples): naive fp32 VALU path.
// ---------------------------------------------------------------------------
__global__ __launch_bounds__(256)
void naive_argmax(const float* __restrict__ e, const float* __restrict__ t,
                  float* __restrict__ out, int D, int V)
{
  const int p   = blockIdx.x;
  const int tid = threadIdx.x;
  const float* ep = e + (long)p * D;
  float bv = -3.0e38f; int bi = 0;
  for (int v = tid; v < V; v += 256) {
    const float* tv = t + (long)v * D;
    float dot = 0.0f, nt = 0.0f;
    for (int d = 0; d < D; ++d) { float x = tv[d]; dot += ep[d] * x; nt += x * x; }
    float sim = dot / fmaxf(sqrtf(nt), 1e-8f);
    if (sim > bv || (sim == bv && v < bi)) { bv = sim; bi = v; }
  }
  __shared__ float sv[256]; __shared__ int si[256];
  sv[tid] = bv; si[tid] = bi;
  __syncthreads();
  for (int s = 128; s > 0; s >>= 1) {
    if (tid < s) {
      if (sv[tid + s] > sv[tid] || (sv[tid + s] == sv[tid] && si[tid + s] < si[tid])) {
        sv[tid] = sv[tid + s]; si[tid] = si[tid + s];
      }
    }
    __syncthreads();
  }
  if (tid == 0) out[p] = (float)si[0];
}

// ---------------------------------------------------------------------------
extern "C" void kernel_launch(void* const* d_in, const int* in_sizes, int n_in,
                              void* d_out, int out_size, void* d_ws, size_t ws_size,
                              hipStream_t stream) {
  const int D = 4096;
  const float* emb   = (const float*)d_in[0];
  const float* table = (const float*)d_in[1];
  const int P = in_sizes[0] / D;          // 2048
  const int V = in_sizes[1] / D;          // 50257
  float* out = (float*)d_out;

  const int VPAD = ((V + TILE_N - 1) / TILE_N) * TILE_N;

  const size_t szT    = (size_t)VPAD * D * sizeof(bf16_t);
  const size_t szE    = (size_t)P * D * sizeof(bf16_t);
  const size_t szInv  = (size_t)VPAD * sizeof(float);
  const size_t szKeys = (size_t)P * sizeof(unsigned long long);
  const size_t need   = 2 * szT + 2 * szE + szInv + szKeys;

  if (ws_size < need || (P % TILE_M) != 0 || (D % TILE_K) != 0) {
    naive_argmax<<<P, 256, 0, stream>>>(emb, table, out, D, V);
    return;
  }

  char* w = (char*)d_ws;
  bf16_t* tHi = (bf16_t*)w;               w += szT;
  bf16_t* tLo = (bf16_t*)w;               w += szT;
  bf16_t* eHi = (bf16_t*)w;               w += szE;
  bf16_t* eLo = (bf16_t*)w;               w += szE;
  float*  invT = (float*)w;               w += szInv;
  unsigned long long* keys = (unsigned long long*)w;

  convert_rows<<<VPAD, 256, 0, stream>>>(table, tHi, tLo, invT, V, D);
  convert_rows<<<P,    256, 0, stream>>>(emb,   eHi, eLo, nullptr, P, D);
  init_keys<<<(P + 255) / 256, 256, 0, stream>>>(keys, P);

  dim3 grid(P / TILE_M, VPAD / TILE_N);   // x fastest: P-tiles sharing a V-tile run adjacently
  wmma_gemm_argmax<<<grid, GEMM_THREADS, 0, stream>>>(eHi, eLo, tHi, tLo, invT, keys, D, V);

  finalize_ids<<<(P + 255) / 256, 256, 0, stream>>>(keys, out, P);
}
